// DGNL_7687991460166
// MI455X (gfx1250) — matrix operand order repaired and checked by reading, not compiled, not run on hardware
//
#include <hip/hip_runtime.h>

#define EPSF 1e-6f

typedef _Float16 h16 __attribute__((ext_vector_type(16)));
typedef _Float16 h8  __attribute__((ext_vector_type(8)));
typedef float    f8  __attribute__((ext_vector_type(8)));

// ---------- WMMA helpers ----------
static __device__ __forceinline__ f8 wmma16(h16 a, h16 b, f8 c) {
  // v_wmma_f32_16x16x32_f16
  return __builtin_amdgcn_wmma_f32_16x16x32_f16(false, a, false, b, (short)0, c,
                                                false, false);
}

// A-fragment (16-bit A 16x32): this lane holds halves p[0..15]
static __device__ __forceinline__ h16 load_a_h(const _Float16* p) {
  h8 a = *(const h8*)p;
  h8 b = *(const h8*)(p + 8);
  h16 r;
#pragma unroll
  for (int i = 0; i < 8; ++i) { r[i] = a[i]; r[i + 8] = b[i]; }
  return r;
}

// A-fragment from f32 weights (convert on load)
static __device__ __forceinline__ h16 load_a_f32cvt(const float* p) {
  h16 r;
#pragma unroll
  for (int i = 0; i < 8; ++i) {
    r[i]     = (_Float16)p[i];
    r[i + 8] = (_Float16)p[i + 8];
  }
  return r;
}

// ---------- problem dims (fixed by reference) ----------
// n=4 c=128 h=w=256 ch=64 ; N=4096 (64x64) ; M=1024 (32x32)

// ---------- K1: depthwise 4x4 stride-4 downsample -> f16 [n][4096][128] ----------
__global__ void k_downsample(const float* __restrict__ x,
                             const float* __restrict__ wdown,
                             _Float16* __restrict__ xd) {
  int idx = blockIdx.x * blockDim.x + threadIdx.x;   // n*128*4096 = 2^21
  if (idx >= (4 << 19)) return;
  int pos = idx & 4095;
  int ch  = (idx >> 12) & 127;
  int n   = idx >> 19;
  int oy = pos >> 6, ox = pos & 63;
  const float* xp = x + (((size_t)n * 128 + ch) * 256 + (size_t)oy * 4) * 256 + ox * 4;
  const float* wp = wdown + ch * 16;
  float s = 0.f;
#pragma unroll
  for (int i = 0; i < 4; ++i)
#pragma unroll
    for (int j = 0; j < 4; ++j) s += xp[i * 256 + j] * wp[i * 4 + j];
  xd[((size_t)n * 4096 + pos) * 128 + ch] = (_Float16)s;
}

// ---------- K2: three 1x1 convs as 64x128 GEMMs via WMMA ----------
// out layout: [n][pos(4096)][ch(64)] f16
__global__ __launch_bounds__(128) void k_conv3(
    const _Float16* __restrict__ xd,
    const float* __restrict__ wtheta, const float* __restrict__ btheta,
    const float* __restrict__ wphi,   const float* __restrict__ bphi,
    const float* __restrict__ wg,     const float* __restrict__ bg,
    _Float16* __restrict__ theta, _Float16* __restrict__ phiF,
    _Float16* __restrict__ gF) {
  int which = blockIdx.z;
  const float* W = (which == 0) ? wtheta : (which == 1) ? wphi : wg;
  const float* B = (which == 0) ? btheta : (which == 1) ? bphi : bg;
  _Float16* O    = (which == 0) ? theta  : (which == 1) ? phiF : gF;
  int n    = blockIdx.y;
  int lane = threadIdx.x & 31;
  int mt   = threadIdx.x >> 5;   // wave = m-tile (ch_out tile) 0..3
  int l16  = lane & 15;
  int hi   = lane >> 4;

  // preload weight A-fragments, K=128 -> 4 chunks of 32
  const float* wr = W + (size_t)(mt * 16 + l16) * 128 + hi * 16;
  h16 A[4];
#pragma unroll
  for (int kc = 0; kc < 4; ++kc) A[kc] = load_a_f32cvt(wr + kc * 32);
  float bias[8];
#pragma unroll
  for (int r = 0; r < 8; ++r) bias[r] = B[mt * 16 + hi * 8 + r];

  for (int t = 0; t < 16; ++t) {
    int nb  = (blockIdx.x * 16 + t) * 16;
    int pos = nb + l16;
    const _Float16* bp = xd + ((size_t)n * 4096 + pos) * 128 + hi * 16;
    f8 C = {};
#pragma unroll
    for (int kc = 0; kc < 4; ++kc)
      C = wmma16(A[kc], *(const h16*)(bp + kc * 32), C);
    h8 o;
#pragma unroll
    for (int r = 0; r < 8; ++r) o[r] = (_Float16)(C[r] + bias[r]);
    *(h8*)(O + ((size_t)n * 4096 + pos) * 64 + mt * 16 + hi * 8) = o;
  }
}

// ---------- K2b: 2x2 maxpool for phi/g ----------
// phiP: [n][opos(1024)][64]  (B-major for matmul-1)
// gP  : [n][ch(64)][opos(1024)] (B-major for matmul-2)
__global__ void k_maxpool(const _Float16* __restrict__ phiF,
                          const _Float16* __restrict__ gF,
                          _Float16* __restrict__ phiP,
                          _Float16* __restrict__ gP) {
  int idx = blockIdx.x * blockDim.x + threadIdx.x;   // 4*1024*64 = 2^18
  if (idx >= (4 << 16)) return;
  int ch = idx & 63;
  int op = (idx >> 6) & 1023;
  int n  = idx >> 16;
  int oy = op >> 5, ox = op & 31;
  int p0 = (oy * 2) * 64 + ox * 2;
  size_t base = (size_t)n * 4096 * 64 + ch;
  float a0 = (float)phiF[base + (size_t)p0 * 64];
  float a1 = (float)phiF[base + (size_t)(p0 + 1) * 64];
  float a2 = (float)phiF[base + (size_t)(p0 + 64) * 64];
  float a3 = (float)phiF[base + (size_t)(p0 + 65) * 64];
  phiP[((size_t)n * 1024 + op) * 64 + ch] =
      (_Float16)fmaxf(fmaxf(a0, a1), fmaxf(a2, a3));
  float b0 = (float)gF[base + (size_t)p0 * 64];
  float b1 = (float)gF[base + (size_t)(p0 + 1) * 64];
  float b2 = (float)gF[base + (size_t)(p0 + 64) * 64];
  float b3 = (float)gF[base + (size_t)(p0 + 65) * 64];
  gP[((size_t)n * 64 + ch) * 1024 + op] =
      (_Float16)fmaxf(fmaxf(b0, b1), fmaxf(b2, b3));
}

// ---------- K3: bilinear (align_corners) depth resizes ----------
static __device__ __forceinline__ float bilerp_depth(const float* dm, int n,
                                                     float fy, float fx) {
  int y0 = (int)fy, x0 = (int)fx;
  float wy = fy - y0, wx = fx - x0;
  int y1 = min(y0 + 1, 255), x1 = min(x0 + 1, 255);
  const float* p = dm + (size_t)n * 65536;
  float r0 = p[y0 * 256 + x0] * (1.f - wx) + p[y0 * 256 + x1] * wx;
  float r1 = p[y1 * 256 + x0] * (1.f - wx) + p[y1 * 256 + x1] * wx;
  return r0 * (1.f - wy) + r1 * wy;
}

__global__ void k_depth(const float* __restrict__ dm, float* __restrict__ d1,
                        float* __restrict__ d2) {
  int idx = blockIdx.x * blockDim.x + threadIdx.x;
  if (idx < 4 * 4096) {                   // 64x64 resize
    int p = idx & 4095, n = idx >> 12;
    float fy = (p >> 6) * (255.f / 63.f);
    float fx = (p & 63) * (255.f / 63.f);
    d1[idx] = bilerp_depth(dm, n, fy, fx);
  } else if (idx < 4 * 4096 + 4 * 1024) { // 32x32 resize
    int j = idx - 4 * 4096;
    int p = j & 1023, n = j >> 10;
    float fy = (p >> 5) * (255.f / 31.f);
    float fx = (p & 31) * (255.f / 31.f);
    d2[j] = bilerp_depth(dm, n, fy, fx);
  }
}

// ---------- K4: fused attention (Ra softmax, Rd softmax, S softmax, S@g) ----------
// 1 block (4 waves) per (n, 16-row tile). LDS: 16x1024 f32 logits + 16x1024 f16 S.
__global__ __launch_bounds__(128) void k_attn(
    const _Float16* __restrict__ theta, const _Float16* __restrict__ phiP,
    const _Float16* __restrict__ gP, const float* __restrict__ d1,
    const float* __restrict__ d2, _Float16* __restrict__ yh) {
  extern __shared__ char smem[];
  float*    logits = (float*)smem;                       // 65536 B
  _Float16* Sh     = (_Float16*)(smem + 65536);          // 32768 B
  float*    red    = (float*)(smem + 65536 + 32768);     // 2*16*8 floats
  float*    rstat  = red + 256;                          // 64 floats
  float*    d2s    = rstat + 64;                         // 1024 floats

  int n       = blockIdx.x >> 8;
  int rowbase = (blockIdx.x & 255) * 16;
  int lane = threadIdx.x & 31;
  int wv   = threadIdx.x >> 5;
  int l16  = lane & 15;
  int hi   = lane >> 4;

  for (int i = threadIdx.x; i < 1024; i += 128) d2s[i] = d2[n * 1024 + i];

  // ---- phase 1: logits = theta_tile (16x64) @ phi (64x1024) ----
  const _Float16* ap = theta + ((size_t)n * 4096 + rowbase + l16) * 64 + hi * 16;
  h16 A0 = load_a_h(ap);
  h16 A1 = load_a_h(ap + 32);
  for (int t = 0; t < 16; ++t) {
    int jb = (wv * 16 + t) * 16;
    const _Float16* bp = phiP + ((size_t)n * 1024 + jb + l16) * 64 + hi * 16;
    f8 C = {};
    C = wmma16(A0, *(const h16*)bp, C);
    C = wmma16(A1, *(const h16*)(bp + 32), C);
    int col = jb + l16;
#pragma unroll
    for (int r = 0; r < 8; ++r) logits[(r + hi * 8) * 1024 + col] = C[r];
  }
  __syncthreads();

  // ---- phase 2: chained softmaxes, 8 threads per row ----
  int row = threadIdx.x >> 3;
  int sub = threadIdx.x & 7;
  float* lrow = logits + row * 1024;

  float m = -1e30f;
  for (int i = 0; i < 128; ++i) m = fmaxf(m, lrow[sub * 128 + i]);
  red[row * 8 + sub] = m;
  __syncthreads();
  if (sub == 0) {
    float mm = red[row * 8];
    for (int j = 1; j < 8; ++j) mm = fmaxf(mm, red[row * 8 + j]);
    rstat[row] = mm;
  }
  __syncthreads();
  float rmax = rstat[row];
  float dd1  = d1[n * 4096 + rowbase + row];

  float sp = 0.f, sq = 0.f;
  for (int i = 0; i < 128; ++i) {
    int cc = sub * 128 + i;
    sp += __expf(lrow[cc] - rmax);
    float dj = d2s[cc];
    float q  = fminf(dd1 / (dj + EPSF), dj / (dd1 + EPSF));
    sq += __expf(q);
  }
  red[row * 8 + sub]       = sp;
  red[128 + row * 8 + sub] = sq;
  __syncthreads();
  if (sub == 0) {
    float s0 = 0.f, s1 = 0.f;
    for (int j = 0; j < 8; ++j) { s0 += red[row * 8 + j]; s1 += red[128 + row * 8 + j]; }
    rstat[16 + row] = s0;
    rstat[32 + row] = s1;
  }
  __syncthreads();
  float sumP = rstat[16 + row];
  float sumQ = rstat[32 + row];

  float st = 0.f;
  for (int i = 0; i < 128; ++i) {
    int cc = sub * 128 + i;
    float p  = __expf(lrow[cc] - rmax) / sumP;
    float dj = d2s[cc];
    float q  = __expf(fminf(dd1 / (dj + EPSF), dj / (dd1 + EPSF))) / sumQ;
    float tt = __expf(p * q);
    lrow[cc] = tt;
    st += tt;
  }
  __syncthreads();          // protect red reuse
  red[row * 8 + sub] = st;
  __syncthreads();
  if (sub == 0) {
    float s0 = 0.f;
    for (int j = 0; j < 8; ++j) s0 += red[row * 8 + j];
    rstat[48 + row] = s0;
  }
  __syncthreads();
  float sumT = rstat[48 + row];
  for (int i = 0; i < 128; ++i) {
    int cc = sub * 128 + i;
    Sh[row * 1024 + cc] = (_Float16)(lrow[cc] / sumT);
  }
  __syncthreads();

  // ---- phase 3: y_tile (16x64) = S (16x1024) @ g (1024x64), wave = 16-ch tile ----
  int chb = wv * 16;
  const _Float16* gbase = gP + ((size_t)n * 64 + chb + l16) * 1024 + hi * 16;
  const _Float16* sbase = Sh + (size_t)l16 * 1024 + hi * 16;
  f8 C = {};
#pragma unroll 4
  for (int kb = 0; kb < 32; ++kb) {
    h16 Av = load_a_h(sbase + kb * 32);            // ds_read_b128 x2
    h16 Bv = *(const h16*)(gbase + kb * 32);
    C = wmma16(Av, Bv, C);
  }
#pragma unroll
  for (int r = 0; r < 8; ++r) {
    int pos = rowbase + r + hi * 8;
    yh[((size_t)n * 4096 + pos) * 64 + chb + l16] = (_Float16)C[r];
  }
}

// ---------- K5a: z = w_z(128x64) @ y + b_z  -> f32 [n][128][4096] ----------
__global__ __launch_bounds__(256) void k_convz(const _Float16* __restrict__ yh,
                                               const float* __restrict__ wz,
                                               const float* __restrict__ bz,
                                               float* __restrict__ z) {
  int n    = blockIdx.y;
  int lane = threadIdx.x & 31;
  int mt   = threadIdx.x >> 5;   // 8 waves -> 8 ch_out tiles
  int l16  = lane & 15;
  int hi   = lane >> 4;

  const float* wr = wz + (size_t)(mt * 16 + l16) * 64 + hi * 16;
  h16 A0 = load_a_f32cvt(wr);
  h16 A1 = load_a_f32cvt(wr + 32);
  float bias[8];
#pragma unroll
  for (int r = 0; r < 8; ++r) bias[r] = bz[mt * 16 + hi * 8 + r];

  for (int t = 0; t < 16; ++t) {
    int posb = (blockIdx.x * 16 + t) * 16;
    int pos  = posb + l16;
    const _Float16* bp = yh + ((size_t)n * 4096 + pos) * 64 + hi * 16;
    f8 C = {};
    C = wmma16(A0, *(const h16*)bp, C);
    C = wmma16(A1, *(const h16*)(bp + 32), C);
#pragma unroll
    for (int r = 0; r < 8; ++r) {
      z[((size_t)n * 128 + mt * 16 + hi * 8 + r) * 4096 + posb + l16] =
          C[r] + bias[r];
    }
  }
}

// ---------- K5b: out = x + bilinear_up4_ac(z) ----------
__global__ void k_out(const float* __restrict__ x, const float* __restrict__ z,
                      float* __restrict__ out) {
  size_t idx = (size_t)blockIdx.x * blockDim.x + threadIdx.x;  // 2^25
  if (idx >= ((size_t)4 << 23)) return;
  int X  = idx & 255;
  int Y  = (idx >> 8) & 255;
  int cc = (idx >> 16) & 127;
  int n  = (int)(idx >> 23);
  float fy = Y * (63.f / 255.f);
  float fx = X * (63.f / 255.f);
  int y0 = (int)fy, x0 = (int)fx;
  float wy = fy - y0, wx = fx - x0;
  int y1 = min(y0 + 1, 63), x1 = min(x0 + 1, 63);
  const float* zp = z + ((size_t)n * 128 + cc) * 4096;
  float v00 = zp[y0 * 64 + x0], v01 = zp[y0 * 64 + x1];
  float v10 = zp[y1 * 64 + x0], v11 = zp[y1 * 64 + x1];
  float v = (v00 * (1.f - wx) + v01 * wx) * (1.f - wy) +
            (v10 * (1.f - wx) + v11 * wx) * wy;
  out[idx] = x[idx] + v;
}

// ---------- launch ----------
extern "C" void kernel_launch(void* const* d_in, const int* in_sizes, int n_in,
                              void* d_out, int out_size, void* d_ws,
                              size_t ws_size, hipStream_t stream) {
  (void)in_sizes; (void)n_in; (void)out_size; (void)ws_size;
  const float* x      = (const float*)d_in[0];
  const float* dmap   = (const float*)d_in[1];
  const float* wtheta = (const float*)d_in[2];
  const float* btheta = (const float*)d_in[3];
  const float* wphi   = (const float*)d_in[4];
  const float* bphi   = (const float*)d_in[5];
  const float* wg     = (const float*)d_in[6];
  const float* bg     = (const float*)d_in[7];
  const float* wdown  = (const float*)d_in[8];
  const float* wz     = (const float*)d_in[9];
  const float* bz     = (const float*)d_in[10];
  float* out = (float*)d_out;

  char* ws = (char*)d_ws;
  _Float16* xd    = (_Float16*)(ws);                    // 4*4096*128*2 = 4 MiB
  _Float16* theta = (_Float16*)(ws + (4u << 20));       // 2 MiB
  _Float16* phiF  = (_Float16*)(ws + (6u << 20));       // 2 MiB
  _Float16* gF    = (_Float16*)(ws + (8u << 20));       // 2 MiB
  _Float16* phiP  = (_Float16*)(ws + (10u << 20));      // 512 KiB
  _Float16* gP    = (_Float16*)(ws + (10u << 20) + (512u << 10));
  float*    d1    = (float*)(ws + (11u << 20));         // 64 KiB
  float*    d2    = (float*)(ws + (11u << 20) + (64u << 10));   // 16 KiB
  _Float16* yh    = (_Float16*)(ws + (12u << 20));      // 2 MiB
  float*    z     = (float*)(ws + (14u << 20));         // 8 MiB

  k_downsample<<<(4 << 19) / 256, 256, 0, stream>>>(x, wdown, xd);
  k_conv3<<<dim3(16, 4, 3), 128, 0, stream>>>(xd, wtheta, btheta, wphi, bphi,
                                              wg, bg, theta, phiF, gF);
  k_maxpool<<<(4 << 16) / 256, 256, 0, stream>>>(phiF, gF, phiP, gP);
  k_depth<<<(4 * 4096 + 4 * 1024 + 255) / 256, 256, 0, stream>>>(dmap, d1, d2);
  const size_t attn_smem = 65536 + 32768 + 256 * 4 + 64 * 4 + 1024 * 4;
  k_attn<<<1024, 128, attn_smem, stream>>>(theta, phiP, gP, d1, d2, yh);
  k_convz<<<dim3(16, 4), 256, 0, stream>>>(yh, wz, bz, z);
  k_out<<<(unsigned)(((size_t)4 << 23) / 256), 256, 0, stream>>>(x, z, out);
}
